// ResidualVectorQuantizer_63917703299729
// MI455X (gfx1250) — matrix operand order repaired and checked by reading, not compiled, not run on hardware
//
#include <hip/hip_runtime.h>

#define NUM_CB 8
#define CB_SIZE 1024
#define DIM 256
#define ROWS 8192            // BATCH * SEQ
#define M_TILE 64
#define N_TILE 32
#define NTILES (CB_SIZE / N_TILE)          // 32
#define BLOCKS_PER_LAYER (ROWS / M_TILE)   // 128
#define OUT_ELEMS (ROWS * DIM)             // 2097152
#define CB_ELEMS (NUM_CB * CB_SIZE * DIM)  // 2097152
#define TILE_ELEMS (N_TILE * DIM)          // 8192 bf16 elems = 16 KB

typedef __attribute__((ext_vector_type(16))) __bf16 v16bf;
typedef __attribute__((ext_vector_type(8)))  float  v8f;
typedef int v4i __attribute__((vector_size(16)));

struct __attribute__((aligned(16))) U128 { unsigned int x, y, z, w; };
union ABu { U128 q[2]; v16bf v; };

#define AS1 __attribute__((address_space(1)))
#define AS3 __attribute__((address_space(3)))

#if defined(__has_builtin)
#if __has_builtin(__builtin_amdgcn_global_load_async_to_lds_b128)
#define HAVE_ASYNC 1
#endif
#endif
#ifndef HAVE_ASYNC
#define HAVE_ASYNC 0
#endif

__device__ __forceinline__ void wait_async_zero() {
#if HAVE_ASYNC
#if __has_builtin(__builtin_amdgcn_s_wait_asynccnt)
    __builtin_amdgcn_s_wait_asynccnt(0);
#else
    asm volatile("s_wait_asynccnt 0x0" ::: "memory");
#endif
#endif
}

// Stage one 32-code bf16 tile (16 KB): 64 B per thread, 4 x 128-bit transfers.
__device__ __forceinline__ void stage_b_tile(const unsigned short* __restrict__ gsrc,
                                             unsigned short* ldst, int tid) {
#if HAVE_ASYNC
    // AS1 pointer value == flat global address; AS3 pointer value == low 32 bits
    // of the flat shared address (LDS offset). Build both via inttoptr.
    AS1 v4i* g = (AS1 v4i*)(unsigned long long)(uintptr_t)(gsrc + tid * 32);
    AS3 v4i* l = (AS3 v4i*)(unsigned int)(uintptr_t)(ldst + tid * 32);
    __builtin_amdgcn_global_load_async_to_lds_b128(g, l, 0,  0);
    __builtin_amdgcn_global_load_async_to_lds_b128(g, l, 16, 0);
    __builtin_amdgcn_global_load_async_to_lds_b128(g, l, 32, 0);
    __builtin_amdgcn_global_load_async_to_lds_b128(g, l, 48, 0);
#else
    #pragma unroll
    for (int i = 0; i < 4; ++i)
        *(uint4*)&ldst[tid * 32 + i * 8] = *(const uint4*)&gsrc[tid * 32 + i * 8];
#endif
}

__device__ inline unsigned short f2bf(float f) {
    unsigned int u = __float_as_uint(f);
    u += 0x7FFFu + ((u >> 16) & 1u);   // round-to-nearest-even
    return (unsigned short)(u >> 16);
}

// ---------------- init: residual = x, out = 0 ----------------
__global__ void vq_init(const float* __restrict__ x,
                        float* __restrict__ residual,
                        float* __restrict__ outbuf, int n) {
    int i = blockIdx.x * blockDim.x + threadIdx.x;
    if (i < n) { residual[i] = x[i]; outbuf[i] = 0.0f; }
}

// ---------------- one-time codebook f32 -> bf16 conversion ----------------
__global__ void vq_cvt(const float* __restrict__ cb,
                       unsigned short* __restrict__ cb_bf) {
    int i = (blockIdx.x * blockDim.x + threadIdx.x) * 4;
    float4 v = *(const float4*)&cb[i];
    ushort4 h;
    h.x = f2bf(v.x); h.y = f2bf(v.y); h.z = f2bf(v.z); h.w = f2bf(v.w);
    *(ushort4*)&cb_bf[i] = h;
}

// ---------------- per-code squared norms (exact f32) ----------------
__global__ void vq_norms(const float* __restrict__ codebooks,
                         float* __restrict__ enorm) {
    int c = blockIdx.x * blockDim.x + threadIdx.x;   // 0 .. 8191
    const float4* p = (const float4*)&codebooks[(size_t)c * DIM];
    float s = 0.0f;
    #pragma unroll 8
    for (int i = 0; i < DIM / 4; ++i) {
        float4 v = p[i];
        s += v.x * v.x + v.y * v.y + v.z * v.z + v.w * v.w;
    }
    enorm[c] = s;
}

// ---------------- one VQ layer: dist GEMM (WMMA bf16) + argmin + update ----------------
__global__ __launch_bounds__(256) void vq_layer(
    const float* __restrict__ codebooks,         // (8,1024,256) f32 (gather/update)
    const unsigned short* __restrict__ cb_bf,    // (8,1024,256) bf16 copy
    const float* __restrict__ enorm,             // (8,1024)
    float* __restrict__ residual,                // (8192,256)
    float* __restrict__ outbuf,                  // (8192,256) accumulator
    float* __restrict__ partials,                // (8,128)
    int layer)
{
    // 2 x 16 KB B double-buffer; also (exactly) holds the 64x256 bf16 A tile
    // during the prologue before being recycled for B.
    __shared__ unsigned short Sbuf[2][TILE_ELEMS];
    __shared__ float bestVs[M_TILE][2];
    __shared__ int   bestIs[M_TILE][2];
    __shared__ int   idxS[M_TILE];
    __shared__ float red[256];

    const int tid   = threadIdx.x;
    const int lane  = tid & 31;
    const int wave  = tid >> 5;          // 0..7
    const int mwave = wave & 3;          // row sub-tile 0..3
    const int nwave = wave >> 2;         // col sub-tile 0..1
    const int half  = lane >> 4;         // lane half selects K slice
    const int l16   = lane & 15;
    const size_t rowBase = (size_t)blockIdx.x * M_TILE;
    const float*          cbL = codebooks + (size_t)layer * CB_SIZE * DIM;
    const unsigned short* cbB = cb_bf     + (size_t)layer * CB_SIZE * DIM;
    const float*          enL = enorm + layer * CB_SIZE;

    // ---- stage residual tile as bf16 into LDS (Sbuf used flat: 16384 elems) ----
    unsigned short* Aflat = &Sbuf[0][0];
    #pragma unroll
    for (int i = 0; i < 16; ++i) {
        int lin = tid * 64 + i * 4;                        // covers 64*256 elems
        float4 v = *(const float4*)&residual[rowBase * DIM + lin];
        ushort4 h;
        h.x = f2bf(v.x); h.y = f2bf(v.y); h.z = f2bf(v.z); h.w = f2bf(v.w);
        *(ushort4*)&Aflat[lin] = h;
    }
    __syncthreads();

    // ---- hoist A fragments into registers (invariant across all 32 B tiles) ----
    const unsigned short* Arow = &Aflat[(mwave * 16 + l16) * DIM];
    ABu aReg[8];
    #pragma unroll
    for (int kc = 0; kc < DIM / 32; ++kc) {
        const int k0 = kc * 32;
        aReg[kc].q[0] = *(const U128*)&Arow[k0 + half * 8];
        aReg[kc].q[1] = *(const U128*)&Arow[k0 + 16 + half * 8];
    }
    __syncthreads();   // all waves done reading A staging; Sbuf free for B

    // ---- prologue: kick off async copy of B tile 0 into Sbuf[0] ----
    stage_b_tile(cbB, &Sbuf[0][0], tid);
    float enCur = enL[nwave * 16 + l16];

    float bestV[8];
    int   bestI[8];
    #pragma unroll
    for (int r = 0; r < 8; ++r) { bestV[r] = 3.4e38f; bestI[r] = 0x7FFFFFFF; }

    for (int nt = 0; nt < NTILES; ++nt) {
        wait_async_zero();   // own async stores into Sbuf[nt&1] have landed in LDS
        __syncthreads();     // every wave's tile-nt data visible; buf[(nt+1)&1] free

        // issue DMA for next tile while this tile computes
        if (nt + 1 < NTILES)
            stage_b_tile(cbB + (size_t)(nt + 1) * TILE_ELEMS,
                         &Sbuf[(nt + 1) & 1][0], tid);
        float enNext = (nt + 1 < NTILES)
                     ? enL[(nt + 1) * N_TILE + nwave * 16 + l16] : 0.0f;

        // ---- 16x16 tile of R . E^T over K=256 ----
        const unsigned short* Brow = &Sbuf[nt & 1][(nwave * 16 + l16) * DIM];
        v8f acc = {};
        #pragma unroll
        for (int kc = 0; kc < DIM / 32; ++kc) {
            const int k0 = kc * 32;
            ABu b;
            // B (32x16 bf16): lane half selects K offset +8; K 8h..8h+7 then 16+8h..16+8h+7
            b.q[0] = *(const U128*)&Brow[k0 + half * 8];
            b.q[1] = *(const U128*)&Brow[k0 + 16 + half * 8];
            acc = __builtin_amdgcn_wmma_f32_16x16x32_bf16(
                false, aReg[kc].v, false, b.v, (short)0, acc, false, false);
        }

        // ---- distance proxy + running argmin ----
        const int col = nt * N_TILE + nwave * 16 + l16;
        #pragma unroll
        for (int r = 0; r < 8; ++r) {
            float s = enCur - 2.0f * acc[r];
            if (s < bestV[r]) { bestV[r] = s; bestI[r] = col; }
        }
        enCur = enNext;
    }

    // ---- reduce argmin across the 16 column-lanes of each half-wave ----
    #pragma unroll
    for (int r = 0; r < 8; ++r) {
        float bv = bestV[r];
        int   bi = bestI[r];
        #pragma unroll
        for (int m = 8; m > 0; m >>= 1) {      // masks 8/4/2/1 stay within 16-lane halves
            float ov = __shfl_xor(bv, m, 32);
            int   oi = __shfl_xor(bi, m, 32);
            if (ov < bv || (ov == bv && oi < bi)) { bv = ov; bi = oi; }
        }
        if (l16 == 0) {
            int row = mwave * 16 + half * 8 + r;   // C layout: VGPR r -> row r (lanes 0-15) / r+8 (16-31)
            bestVs[row][nwave] = bv;
            bestIs[row][nwave] = bi;
        }
    }
    __syncthreads();
    if (tid < M_TILE) {
        float v0 = bestVs[tid][0], v1 = bestVs[tid][1];
        int   i0 = bestIs[tid][0], i1 = bestIs[tid][1];
        idxS[tid] = (v1 < v0 || (v1 == v0 && i1 < i0)) ? i1 : i0;
    }
    __syncthreads();

    // ---- gather + update residual/out + loss accumulation (exact f32) ----
    const int r  = tid >> 2;            // 64 rows x 4 threads/row
    const int d0 = (tid & 3) * 64;      // 64 dims per thread
    const size_t grow = rowBase + r;
    const int code = idxS[r];
    const float4* qv = (const float4*)&cbL[(size_t)code * DIM + d0];
    float4* rv = (float4*)&residual[grow * DIM + d0];
    float4* ov = (float4*)&outbuf[grow * DIM + d0];
    float err = 0.0f;
    #pragma unroll
    for (int i = 0; i < 16; ++i) {
        float4 q  = qv[i];
        float4 rr = rv[i];
        rr.x -= q.x; rr.y -= q.y; rr.z -= q.z; rr.w -= q.w;
        rv[i] = rr;
        float4 o = ov[i];
        o.x += q.x; o.y += q.y; o.z += q.z; o.w += q.w;
        ov[i] = o;
        err += rr.x * rr.x + rr.y * rr.y + rr.z * rr.z + rr.w * rr.w;
    }
    red[tid] = err;
    __syncthreads();
    #pragma unroll
    for (int s = 128; s > 0; s >>= 1) {
        if (tid < s) red[tid] += red[tid + s];
        __syncthreads();
    }
    if (tid == 0) partials[layer * BLOCKS_PER_LAYER + blockIdx.x] = red[0];
}

// ---------------- deterministic final loss reduction ----------------
__global__ void vq_final_loss(const float* __restrict__ partials,
                              float* __restrict__ loss_out) {
    __shared__ float red[256];
    float s = 0.0f;
    for (int i = threadIdx.x; i < NUM_CB * BLOCKS_PER_LAYER; i += 256)
        s += partials[i];
    red[threadIdx.x] = s;
    __syncthreads();
    for (int st = 128; st > 0; st >>= 1) {
        if (threadIdx.x < st) red[threadIdx.x] += red[threadIdx.x + st];
        __syncthreads();
    }
    // loss per layer = (1 + beta) * mean(r_new^2), beta = 0.25
    if (threadIdx.x == 0)
        *loss_out = red[0] * (1.25f / (float)OUT_ELEMS);
}

extern "C" void kernel_launch(void* const* d_in, const int* in_sizes, int n_in,
                              void* d_out, int out_size, void* d_ws, size_t ws_size,
                              hipStream_t stream) {
    const float* x         = (const float*)d_in[0];   // (8,1024,256)
    const float* codebooks = (const float*)d_in[1];   // (8,1024,256)
    float* outp = (float*)d_out;                      // 2097152 out + 1 loss

    float* ws       = (float*)d_ws;
    float* residual = ws;                               // 2097152 floats
    float* enorm    = ws + OUT_ELEMS;                   // 8192 floats
    float* partials = enorm + NUM_CB * CB_SIZE;         // 1024 floats
    unsigned short* cb_bf = (unsigned short*)(partials + NUM_CB * BLOCKS_PER_LAYER);
    // cb_bf offset = (2097152 + 8192 + 1024) * 4 bytes -> 16-byte aligned

    vq_init<<<(OUT_ELEMS + 255) / 256, 256, 0, stream>>>(x, residual, outp, OUT_ELEMS);
    vq_cvt<<<CB_ELEMS / 4 / 256, 256, 0, stream>>>(codebooks, cb_bf);
    vq_norms<<<(NUM_CB * CB_SIZE) / 256, 256, 0, stream>>>(codebooks, enorm);

    for (int layer = 0; layer < NUM_CB; ++layer) {
        vq_layer<<<BLOCKS_PER_LAYER, 256, 0, stream>>>(
            codebooks, cb_bf, enorm, residual, outp, partials, layer);
    }

    vq_final_loss<<<1, 256, 0, stream>>>(partials, outp + OUT_ELEMS);
}
